// EncoderBlock_65335042506819
// MI455X (gfx1250) — matrix operand order repaired and checked by reading, not compiled, not run on hardware
//
#include <hip/hip_runtime.h>
#include <hip/hip_bf16.h>

typedef __bf16 bf16_t;
typedef __attribute__((ext_vector_type(16))) __bf16 v16bf;
typedef __attribute__((ext_vector_type(8)))  __bf16 v8bf;
typedef __attribute__((ext_vector_type(8)))  float  v8f;
typedef __attribute__((ext_vector_type(4)))  float  v4f;

static constexpr int NRES = 16384, DMODEL = 256, KNB = 32, PDIM = 64, NATOM = 5;
static constexpr int NH = 8, KDH = 32, NPTS = 8, DFF = 512;
static constexpr int FINP = 95, FPAD = 96, FPAIR = 162, FPPAD = 192, FOUTC = 1024;
static constexpr int RELCLIP = 32;
#define RBF_SIGMA 0.625f
#define RBF_STEP  (10.0f / 15.0f)
#define USE_ASYNC 1

// ---------------- helpers ----------------
__device__ inline bf16_t tobf(float f) { return (bf16_t)f; }   // RNE fptrunc
__device__ inline float gelu_t(float x) {
  float x3 = x * x * x;
  return 0.5f * x * (1.0f + tanhf(0.7978845608f * (x + 0.044715f * x3)));
}
__device__ inline float wred_max(float v) {
  for (int s = 16; s > 0; s >>= 1) v = fmaxf(v, __shfl_xor(v, s, 32));
  return v;
}
__device__ inline float wred_sum(float v) {
  for (int s = 16; s > 0; s >>= 1) v += __shfl_xor(v, s, 32);
  return v;
}
__device__ inline v16bf cat16(v8bf lo, v8bf hi) {
  return __builtin_shufflevector(lo, hi, 0,1,2,3,4,5,6,7,8,9,10,11,12,13,14,15);
}
#if USE_ASYNC
// 16-byte global -> LDS async copy (ASYNCcnt-tracked, CDNA5 ISA)
__device__ inline void async_b128(void* lds_dst, const void* gsrc) {
  unsigned lds_off = (unsigned)(unsigned long long)lds_dst;  // LDS aperture: addr[31:0]
  asm volatile("global_load_async_to_lds_b128 %0, %1, off"
               :: "v"(lds_off), "v"(gsrc) : "memory");
}
__device__ inline void wait_async0() {
  asm volatile("s_wait_asynccnt 0" ::: "memory");
}
#endif

// ---------------- frames + node features ----------------
__global__ void frames_kernel(const float* __restrict__ pos, float* __restrict__ R,
                              float* __restrict__ LP, float* __restrict__ FEAT) {
  int n = blockIdx.x * blockDim.x + threadIdx.x;
  if (n >= NRES) return;
  const float* p = pos + (size_t)n * NATOM * 3;
  float Nv[3], CA[3], Cv[3];
  for (int i = 0; i < 3; i++) { Nv[i] = p[i]; CA[i] = p[3 + i]; Cv[i] = p[6 + i]; }
  float e1[3], e2[3], e3[3], u[3];
  float s1 = 1e-6f;
  for (int i = 0; i < 3; i++) { e1[i] = Cv[i] - CA[i]; s1 += e1[i] * e1[i]; }
  float inv1 = rsqrtf(s1);
  for (int i = 0; i < 3; i++) e1[i] *= inv1;
  float du = 0.f;
  for (int i = 0; i < 3; i++) { u[i] = Nv[i] - CA[i]; du += u[i] * e1[i]; }
  float s2 = 1e-6f;
  for (int i = 0; i < 3; i++) { e2[i] = u[i] - du * e1[i]; s2 += e2[i] * e2[i]; }
  float inv2 = rsqrtf(s2);
  for (int i = 0; i < 3; i++) e2[i] *= inv2;
  e3[0] = e1[1] * e2[2] - e1[2] * e2[1];
  e3[1] = e1[2] * e2[0] - e1[0] * e2[2];
  e3[2] = e1[0] * e2[1] - e1[1] * e2[0];
  float* Rn = R + (size_t)n * 9;
  for (int i = 0; i < 3; i++) { Rn[i*3+0] = e1[i]; Rn[i*3+1] = e2[i]; Rn[i*3+2] = e3[i]; }
  float* lp = LP + (size_t)n * 15;
  float* f  = FEAT + (size_t)n * FPAD;
  for (int a = 0; a < NATOM; a++) {
    float d[3]; for (int i = 0; i < 3; i++) d[i] = p[a*3+i] - CA[i];
    float ss = 0.f, l[3];
    for (int i = 0; i < 3; i++) {
      l[i] = Rn[0*3+i]*d[0] + Rn[1*3+i]*d[1] + Rn[2*3+i]*d[2];
      lp[a*3+i] = l[i]; ss += l[i]*l[i];
    }
    float invn = rsqrtf(ss + 1e-6f);
    for (int i = 0; i < 3; i++) f[a*3+i] = l[i] * invn;
    float dl = sqrtf(ss + 1e-12f);
    for (int b = 0; b < 16; b++) {
      float z = (dl - RBF_STEP * b) / RBF_SIGMA;
      f[15 + a*16 + b] = __expf(-z*z);
    }
  }
  f[95] = 0.f;
}

// ---------------- generic WMMA GEMM: C = A@B (+bias)(+resid) ----------------
// Block tile 128x64, 8 waves, each wave 32x32 = 4 x v_wmma per K-step.
// A: M x Kd row-major (lda), B: Kd x Nd row-major (ldb). B staged TRANSPOSED in LDS.
__global__ void gemm_wmma_kernel(const float* __restrict__ A, int lda,
                                 const float* __restrict__ B, int ldb,
                                 const float* __restrict__ bias,
                                 const float* __restrict__ resid,
                                 float* __restrict__ C, float* __restrict__ C2,
                                 int M, int Kd, int Nd) {
  __shared__ alignas(16) bf16_t As[128][40];  // rows 80B, 16B-aligned
  __shared__ alignas(16) bf16_t Bt[64][40];   // B^T: [n][k]
  const int tid = threadIdx.x;
  const int bm = blockIdx.x * 128, bn = blockIdx.y * 64;
  const int wid = tid >> 5, lane = tid & 31;
  const int wm = (wid & 3) * 32, wn = (wid >> 2) * 32;
  v8f c00 = {0,0,0,0,0,0,0,0}, c01 = c00, c10 = c00, c11 = c00;
  const bf16_t z = tobf(0.f);
  const int ksteps = (Kd + 31) / 32;
  const int mr0 = wm + (lane & 15), mr1 = mr0 + 16;
  const int kb  = (lane < 16) ? 0 : 8;
  const int kb2 = (lane < 16) ? 0 : 16;
  const int nc0 = wn + (lane & 15), nc1 = nc0 + 16;
  for (int ks = 0; ks < ksteps; ks++) {
    const int k0 = ks * 32;
    if (k0 + 32 <= Kd) {
      // A: 2 x (8 contiguous fp32 -> packed v8bf store); 128 rows x 4 quadrants
      int r = tid >> 2, kq = (tid & 3) * 8;
      for (int half = 0; half < 2; half++) {
        int rr = r + half * 64;
        const float* ap = &A[(size_t)(bm + rr) * lda + k0 + kq];
        v4f a0 = *(const v4f*)ap, a1 = *(const v4f*)(ap + 4);
        v8f af8 = {a0[0],a0[1],a0[2],a0[3],a1[0],a1[1],a1[2],a1[3]};
        *(v8bf*)&As[rr][kq] = __builtin_convertvector(af8, v8bf);
      }
      // B: column gather (coalesced across n), packed transposed store
      int nn = tid & 63, kg = (tid >> 6) * 8;
      const float* bp = &B[(size_t)(k0 + kg) * ldb + bn + nn];
      v8f bf8;
      for (int e = 0; e < 8; e++) bf8[e] = bp[(size_t)e * ldb];
      *(v8bf*)&Bt[nn][kg] = __builtin_convertvector(bf8, v8bf);
    } else {
      for (int i = tid; i < 128 * 32; i += 256) {
        int r = i >> 5, kk = i & 31, gk = k0 + kk;
        As[r][kk] = (gk < Kd) ? tobf(A[(size_t)(bm + r) * lda + gk]) : z;
      }
      for (int i = tid; i < 64 * 32; i += 256) {
        int nn = i >> 5, kk = i & 31, gk = k0 + kk;
        Bt[nn][kk] = (gk < Kd) ? tobf(B[(size_t)gk * ldb + bn + nn]) : z;
      }
    }
    __syncthreads();
    if (ks + 1 < ksteps) {   // speculative prefetch of next K-slab
      __builtin_prefetch(&A[(size_t)(bm + (tid & 127)) * lda + k0 + 32], 0, 1);
      __builtin_prefetch(&B[(size_t)(k0 + 32 + (tid & 31)) * ldb + bn], 0, 1);
    }
    v16bf a0 = cat16(*(const v8bf*)&As[mr0][kb],  *(const v8bf*)&As[mr0][kb + 16]);
    v16bf a1 = cat16(*(const v8bf*)&As[mr1][kb],  *(const v8bf*)&As[mr1][kb + 16]);
    v16bf b0 = cat16(*(const v8bf*)&Bt[nc0][kb], *(const v8bf*)&Bt[nc0][kb + 16]);
    v16bf b1 = cat16(*(const v8bf*)&Bt[nc1][kb], *(const v8bf*)&Bt[nc1][kb + 16]);
    c00 = __builtin_amdgcn_wmma_f32_16x16x32_bf16(false, a0, false, b0, (short)0, c00, false, false);
    c01 = __builtin_amdgcn_wmma_f32_16x16x32_bf16(false, a0, false, b1, (short)0, c01, false, false);
    c10 = __builtin_amdgcn_wmma_f32_16x16x32_bf16(false, a1, false, b0, (short)0, c10, false, false);
    c11 = __builtin_amdgcn_wmma_f32_16x16x32_bf16(false, a1, false, b1, (short)0, c11, false, false);
    __syncthreads();
  }
  for (int r = 0; r < 8; r++) {
    int row0 = bm + wm + r + ((lane < 16) ? 0 : 8);
    int row1 = row0 + 16;
    int col0 = bn + wn + (lane & 15), col1 = col0 + 16;
    float v00 = c00[r], v01 = c01[r], v10 = c10[r], v11 = c11[r];
    if (bias) {
      float bc0 = bias[col0], bc1 = bias[col1];
      v00 += bc0; v01 += bc1; v10 += bc0; v11 += bc1;
    }
    if (resid) {
      v00 += resid[(size_t)row0 * Nd + col0]; v01 += resid[(size_t)row0 * Nd + col1];
      v10 += resid[(size_t)row1 * Nd + col0]; v11 += resid[(size_t)row1 * Nd + col1];
    }
    C[(size_t)row0 * Nd + col0] = v00; C[(size_t)row0 * Nd + col1] = v01;
    C[(size_t)row1 * Nd + col0] = v10; C[(size_t)row1 * Nd + col1] = v11;
    if (C2) {
      C2[(size_t)row0 * Nd + col0] = v00; C2[(size_t)row0 * Nd + col1] = v01;
      C2[(size_t)row1 * Nd + col0] = v10; C2[(size_t)row1 * Nd + col1] = v11;
    }
  }
}

// ---------------- layer norm (one wave per row of 256) ----------------
__global__ void ln_kernel(const float* __restrict__ in, const float* __restrict__ sc,
                          const float* __restrict__ of, float* __restrict__ out) {
  int row = blockIdx.x * 8 + (threadIdx.x >> 5);
  int lane = threadIdx.x & 31;
  const float* x = in + (size_t)row * DMODEL;
  float v[8]; float s = 0.f;
  for (int i = 0; i < 8; i++) { v[i] = x[lane + 32 * i]; s += v[i]; }
  s = wred_sum(s);
  float mu = s / 256.f;
  float vs = 0.f;
  for (int i = 0; i < 8; i++) { float d = v[i] - mu; vs += d * d; }
  vs = wred_sum(vs);
  float inv = rsqrtf(vs / 256.f + 1e-5f);
  float* o = out + (size_t)row * DMODEL;
  for (int i = 0; i < 8; i++) { int c = lane + 32 * i; o[c] = (v[i] - mu) * inv * sc[c] + of[c]; }
}

// ---------------- local->global point transform (in place) ----------------
__global__ void ptglob_kernel(const float* __restrict__ R, const float* __restrict__ pos,
                              float* __restrict__ pts) {
  int idx = blockIdx.x * blockDim.x + threadIdx.x;   // n*64 + h*8 + p
  if (idx >= NRES * NH * NPTS) return;
  int n = idx / (NH * NPTS);
  const float* Rn = R + (size_t)n * 9;
  const float* tn = pos + (size_t)n * 15 + 3;        // CA
  float* p = pts + (size_t)idx * 3;
  float l0 = p[0], l1 = p[1], l2 = p[2];
  for (int i = 0; i < 3; i++)
    p[i] = Rn[i*3+0]*l0 + Rn[i*3+1]*l1 + Rn[i*3+2]*l2 + tn[i];
}

// ---------------- fused pair-feature + WMMA MLP ----------------
__global__ void pair_kernel(const float* __restrict__ pos, const float* __restrict__ R,
                            const int* __restrict__ neigh, const int* __restrict__ resi,
                            const int* __restrict__ chain, const int* __restrict__ batch,
                            const float* __restrict__ w_pair, const float* __restrict__ b_pair,
                            float* __restrict__ pair) {
  __shared__ alignas(16) bf16_t Wt[PDIM][FPPAD + 8];  // w_pair^T: [c][f], rows 400B
  __shared__ alignas(16) bf16_t Pf[KNB][FPPAD + 8];
  __shared__ float relS[KNB][15];
  __shared__ float dijS[KNB][5];
  __shared__ int   rpS[KNB];
  __shared__ float scS[KNB], sbS[KNB];
  const int n = blockIdx.x, tid = threadIdx.x;
  // stage w_pair transposed, packed v8bf stores
  for (int i = tid; i < PDIM * (FPPAD / 8); i += 256) {
    int c = i / (FPPAD / 8), f0 = (i % (FPPAD / 8)) * 8;
    v8f w;
    for (int e = 0; e < 8; e++) {
      int f = f0 + e;
      w[e] = (f < FPAIR) ? w_pair[(size_t)f * PDIM + c] : 0.f;
    }
    *(v8bf*)&Wt[c][f0] = __builtin_convertvector(w, v8bf);
  }
  const float* Rn = R + (size_t)n * 9;
  const float* tn = pos + (size_t)n * 15 + 3;
  if (tid < KNB) {
    int k = tid;
    int nb = neigh[n * KNB + k];
    int j = nb < 0 ? 0 : (nb > NRES - 1 ? NRES - 1 : nb);
    for (int a = 0; a < NATOM; a++) {
      float d[3];
      for (int i = 0; i < 3; i++) d[i] = pos[(size_t)j * 15 + a*3 + i] - tn[i];
      float ss = 0.f;
      for (int i = 0; i < 3; i++) {
        float r = Rn[0*3+i]*d[0] + Rn[1*3+i]*d[1] + Rn[2*3+i]*d[2];
        relS[k][a*3+i] = r; ss += r * r;
      }
      dijS[k][a] = sqrtf(ss + 1e-12f);
    }
    int rp = resi[j] - resi[n];
    rp = (rp < -RELCLIP ? -RELCLIP : (rp > RELCLIP ? RELCLIP : rp)) + RELCLIP;
    rpS[k] = rp;
    scS[k] = (chain[j] == chain[n]) ? 1.f : 0.f;
    sbS[k] = (batch[j] == batch[n]) ? 1.f : 0.f;
  }
  __syncthreads();
  for (int i = tid; i < KNB * FPPAD; i += 256) {
    int k = i / FPPAD, f = i % FPPAD;
    float val = 0.f;
    if (f < 15)       { int a = f / 3; val = relS[k][f] / (dijS[k][a] + 1e-6f); }
    else if (f < 95)  { int a = (f - 15) >> 4; int b = (f - 15) & 15;
                        float zz = (dijS[k][a] - RBF_STEP * b) / RBF_SIGMA; val = __expf(-zz*zz); }
    else if (f < 160) { val = ((f - 95) == rpS[k]) ? 1.f : 0.f; }
    else if (f == 160) val = scS[k];
    else if (f == 161) val = sbS[k];
    Pf[k][f] = tobf(val);
  }
  __syncthreads();
  const int wid = tid >> 5, lane = tid & 31;
  const int wm = (wid & 1) * 16, wn = (wid >> 1) * 16;   // 2x4 wave tiles = 32x64
  const int mrow = wm + (lane & 15);
  const int kb  = (lane < 16) ? 0 : 8;
  const int kb2 = (lane < 16) ? 0 : 16;
  const int nc  = wn + (lane & 15);
  v8f acc = {0,0,0,0,0,0,0,0};
  for (int ks = 0; ks < FPPAD / 32; ks++) {
    const int k0 = ks * 32;
    v16bf af = cat16(*(const v8bf*)&Pf[mrow][k0 + kb],  *(const v8bf*)&Pf[mrow][k0 + kb + 16]);
    v16bf bf = cat16(*(const v8bf*)&Wt[nc][k0 + kb],   *(const v8bf*)&Wt[nc][k0 + kb + 16]);
    acc = __builtin_amdgcn_wmma_f32_16x16x32_bf16(false, af, false, bf, (short)0, acc, false, false);
  }
  for (int r = 0; r < 8; r++) {
    int row = wm + r + ((lane < 16) ? 0 : 8);
    int col = wn + (lane & 15);
    float v = acc[r] + b_pair[col];
    pair[(size_t)n * KNB * PDIM + row * PDIM + col] = gelu_t(v);
  }
}

// ---------------- IPA attention (gather-bound, VALU) ----------------
__global__ void attn_kernel(const float* __restrict__ Q, const float* __restrict__ Kmat,
                            const float* __restrict__ V, const float* __restrict__ QP,
                            const float* __restrict__ KP, const float* __restrict__ VP,
                            const float* __restrict__ pair, const float* __restrict__ wb,
                            const float* __restrict__ gamma,
                            const int* __restrict__ neigh, const int* __restrict__ batch,
                            const unsigned char* __restrict__ mask,
                            const float* __restrict__ R, const float* __restrict__ pos,
                            float* __restrict__ ipa) {
  __shared__ alignas(16) float sq[DMODEL];
  __shared__ alignas(16) float sqp[NH * NPTS * 3];
  __shared__ alignas(16) float pairS[KNB * PDIM];
  __shared__ float attS[KNB * NH];
  __shared__ int   idxS[KNB];
  __shared__ float mk[KNB];
  const int n = blockIdx.x, tid = threadIdx.x;
#if USE_ASYNC
  // pure global->LDS copies: issue async b128 transfers (ASYNCcnt)
  for (int i = tid; i < (KNB * PDIM) / 4; i += 256)
    async_b128(&pairS[i * 4], &pair[(size_t)n * KNB * PDIM + i * 4]);
  if (tid < DMODEL / 4)      async_b128(&sq[tid * 4],  &Q[(size_t)n * DMODEL + tid * 4]);
  if (tid < (NH*NPTS*3) / 4) async_b128(&sqp[tid * 4], &QP[(size_t)n * 192 + tid * 4]);
#else
  for (int i = tid; i < DMODEL; i += 256) sq[i] = Q[(size_t)n * DMODEL + i];
  for (int i = tid; i < NH * NPTS * 3; i += 256) sqp[i] = QP[(size_t)n * 192 + i];
  for (int i = tid; i < KNB * PDIM; i += 256) pairS[i] = pair[(size_t)n * KNB * PDIM + i];
#endif
  if (tid < KNB) {
    int nb = neigh[n * KNB + tid];
    int j = nb < 0 ? 0 : (nb > NRES - 1 ? NRES - 1 : nb);
    idxS[tid] = j;
    bool pm = mask[n] && mask[j] && (nb >= 0) && (batch[j] == batch[n]);
    mk[tid] = pm ? 1.f : 0.f;
  }
#if USE_ASYNC
  wait_async0();
#endif
  __syncthreads();
  const int k = tid & 31, h = tid >> 5;   // wave h owns all 32 neighbours of head h
  {
    int j = idxS[k];
    float qk = 0.f;
    const v4f* q4 = (const v4f*)&sq[h * KDH];
    const v4f* k4 = (const v4f*)&Kmat[(size_t)j * DMODEL + h * KDH];
    for (int e = 0; e < KDH / 4; e++) {
      v4f qv = q4[e], kv = k4[e];
      qk += qv[0]*kv[0] + qv[1]*kv[1] + qv[2]*kv[2] + qv[3]*kv[3];
    }
    qk *= 0.1767766953f;                  // 1/sqrt(32)
    float bterm = 0.f;
    for (int c = 0; c < PDIM; c++) bterm += pairS[k * PDIM + c] * wb[c * NH + h];
    float d2 = 0.f;
    const v4f* qp4 = (const v4f*)&sqp[h * 24];
    const v4f* kp4 = (const v4f*)&KP[(size_t)j * 192 + h * 24];
    for (int e = 0; e < 6; e++) {
      v4f a = qp4[e], b = kp4[e];
      v4f df = {a[0]-b[0], a[1]-b[1], a[2]-b[2], a[3]-b[3]};
      d2 += df[0]*df[0] + df[1]*df[1] + df[2]*df[2] + df[3]*df[3];
    }
    float gm = logf(1.f + __expf(gamma[h]));          // softplus
    const float wc = 0.1666666667f, wl = 0.5773502692f;
    float logit = wl * (qk + bterm - 0.5f * wc * gm * d2);
    if (mk[k] == 0.f) logit = -1e9f;
    float mx = wred_max(logit);
    float ex = __expf(logit - mx);
    float sm = wred_sum(ex);
    attS[k * NH + h] = (mk[k] != 0.f) ? ex / (sm + 1e-20f) : 0.f;
  }
  __syncthreads();
  { // o_s: (h, d) per thread; lanes of wave h read consecutive d -> coalesced per kk
    int dd = tid & 31;
    float acc = 0.f;
    for (int kk = 0; kk < KNB; kk++)
      acc += attS[kk * NH + h] * V[(size_t)idxS[kk] * DMODEL + h * KDH + dd];
    ipa[(size_t)n * FOUTC + h * KDH + dd] = acc;
  }
  { // o_pr
    int lane = tid & 31;
    for (int c = lane; c < PDIM; c += 32) {
      float acc = 0.f;
      for (int kk = 0; kk < KNB; kk++) acc += attS[kk * NH + h] * pairS[kk * PDIM + c];
      ipa[(size_t)n * FOUTC + 256 + h * PDIM + c] = acc;
    }
  }
  if (tid < NH * NPTS) { // o_pt + o_n
    int hh = tid >> 3, p = tid & 7;
    float g0 = 0.f, g1 = 0.f, g2 = 0.f;
    for (int kk = 0; kk < KNB; kk++) {
      float a = attS[kk * NH + hh];
      const float* vp = VP + (size_t)idxS[kk] * 192 + hh * 24 + p * 3;
      g0 += a * vp[0]; g1 += a * vp[1]; g2 += a * vp[2];
    }
    const float* Rn = R + (size_t)n * 9;
    const float* tn = pos + (size_t)n * 15 + 3;
    float d0 = g0 - tn[0], d1 = g1 - tn[1], d2 = g2 - tn[2];
    float ss = 0.f;
    for (int i = 0; i < 3; i++) {
      float lo = Rn[0*3+i]*d0 + Rn[1*3+i]*d1 + Rn[2*3+i]*d2;
      ipa[(size_t)n * FOUTC + 768 + hh * 24 + p * 3 + i] = lo;
      ss += lo * lo;
    }
    ipa[(size_t)n * FOUTC + 960 + hh * 8 + p] = sqrtf(ss + 1e-8f);
  }
}

// ---------------- FFN gate ----------------
__global__ void gelumul_kernel(float* __restrict__ h1, const float* __restrict__ h2, int total) {
  int i = blockIdx.x * blockDim.x + threadIdx.x;
  if (i < total) h1[i] = gelu_t(h1[i]) * h2[i];
}

// ---------------- position update ----------------
__global__ void posupd_kernel(const float* __restrict__ X3, const float* __restrict__ wpos,
                              const float* __restrict__ LP, const float* __restrict__ R,
                              const float* __restrict__ pos, const unsigned char* __restrict__ um,
                              float* __restrict__ pos_out) {
  int n = blockIdx.x * blockDim.x + threadIdx.x;
  if (n >= NRES) return;
  const float* x = X3 + (size_t)n * DMODEL;
  float lp[15];
  for (int o = 0; o < 15; o++) lp[o] = LP[(size_t)n * 15 + o];
  for (int d = 0; d < DMODEL; d++) {
    float xv = x[d];
    const float* wrow = &wpos[d * 15];
    for (int o = 0; o < 15; o++) lp[o] += xv * wrow[o];
  }
  const float* Rn = R + (size_t)n * 9;
  const float* tn = pos + (size_t)n * 15 + 3;
  bool upd = um[n] != 0;
  for (int a = 0; a < NATOM; a++)
    for (int i = 0; i < 3; i++) {
      float np = Rn[i*3+0]*lp[a*3+0] + Rn[i*3+1]*lp[a*3+1] + Rn[i*3+2]*lp[a*3+2] + tn[i];
      pos_out[(size_t)n * 15 + a*3 + i] = upd ? np : pos[(size_t)n * 15 + a*3 + i];
    }
}

// ---------------- host orchestration ----------------
extern "C" void kernel_launch(void* const* d_in, const int* in_sizes, int n_in,
                              void* d_out, int out_size, void* d_ws, size_t ws_size,
                              hipStream_t stream) {
  (void)in_sizes; (void)n_in; (void)out_size; (void)ws_size;
  const float* local  = (const float*)d_in[0];
  const float* pos    = (const float*)d_in[1];
  const int*   neigh  = (const int*)d_in[2];
  const int*   resi   = (const int*)d_in[3];
  const int*   chain  = (const int*)d_in[4];
  const int*   batch  = (const int*)d_in[5];
  const unsigned char* umask = (const unsigned char*)d_in[6];
  const unsigned char* mask  = (const unsigned char*)d_in[7];
  const float* w_in  = (const float*)d_in[8];
  const float* w_pair= (const float*)d_in[9];
  const float* b_pair= (const float*)d_in[10];
  const float* ln1_s = (const float*)d_in[11];
  const float* ln1_o = (const float*)d_in[12];
  const float* wq    = (const float*)d_in[13];
  const float* wk    = (const float*)d_in[14];
  const float* wv    = (const float*)d_in[15];
  const float* wqp   = (const float*)d_in[16];
  const float* wkp   = (const float*)d_in[17];
  const float* wvp   = (const float*)d_in[18];
  const float* wb    = (const float*)d_in[19];
  const float* gamma = (const float*)d_in[20];
  const float* wo    = (const float*)d_in[21];
  const float* bo    = (const float*)d_in[22];
  const float* ln2_s = (const float*)d_in[23];
  const float* ln2_o = (const float*)d_in[24];
  const float* wg    = (const float*)d_in[25];
  const float* wvm   = (const float*)d_in[26];
  const float* wom   = (const float*)d_in[27];
  const float* ln3_s = (const float*)d_in[28];
  const float* ln3_o = (const float*)d_in[29];
  const float* wpos  = (const float*)d_in[30];

  float* ws = (float*)d_ws;
  size_t off = 0;
  auto alloc = [&](size_t nf) { float* p = ws + off; off += nf; return p; };
  float* R    = alloc((size_t)NRES * 9);
  float* LP   = alloc((size_t)NRES * 15);
  float* FEAT = alloc((size_t)NRES * FPAD);
  float* ACC  = alloc((size_t)NRES * DMODEL);
  float* X    = alloc((size_t)NRES * DMODEL);
  float* Qb   = alloc((size_t)NRES * DMODEL);
  float* Kb   = alloc((size_t)NRES * DMODEL);
  float* Vb   = alloc((size_t)NRES * DMODEL);
  float* QP   = alloc((size_t)NRES * 192);
  float* KP   = alloc((size_t)NRES * 192);
  float* VP   = alloc((size_t)NRES * 192);
  float* PAIR = alloc((size_t)NRES * KNB * PDIM);
  float* IPA  = alloc((size_t)NRES * FOUTC);
  float* H1   = alloc((size_t)NRES * DFF);
  float* H2   = alloc((size_t)NRES * DFF);

  float* out_local = (float*)d_out;
  float* out_pos   = out_local + (size_t)NRES * DMODEL;

  dim3 b256(256);
  frames_kernel<<<dim3((NRES + 255) / 256), b256, 0, stream>>>(pos, R, LP, FEAT);
  gemm_wmma_kernel<<<dim3(NRES/128, DMODEL/64), b256, 0, stream>>>(
      FEAT, FPAD, w_in, DMODEL, nullptr, local, ACC, nullptr, NRES, FINP, DMODEL);
  pair_kernel<<<dim3(NRES), b256, 0, stream>>>(pos, R, neigh, resi, chain, batch, w_pair, b_pair, PAIR);
  ln_kernel<<<dim3(NRES/8), b256, 0, stream>>>(ACC, ln1_s, ln1_o, X);
  gemm_wmma_kernel<<<dim3(NRES/128, DMODEL/64), b256, 0, stream>>>(
      X, DMODEL, wq, DMODEL, nullptr, nullptr, Qb, nullptr, NRES, DMODEL, DMODEL);
  gemm_wmma_kernel<<<dim3(NRES/128, DMODEL/64), b256, 0, stream>>>(
      X, DMODEL, wk, DMODEL, nullptr, nullptr, Kb, nullptr, NRES, DMODEL, DMODEL);
  gemm_wmma_kernel<<<dim3(NRES/128, DMODEL/64), b256, 0, stream>>>(
      X, DMODEL, wv, DMODEL, nullptr, nullptr, Vb, nullptr, NRES, DMODEL, DMODEL);
  gemm_wmma_kernel<<<dim3(NRES/128, 192/64), b256, 0, stream>>>(
      X, DMODEL, wqp, 192, nullptr, nullptr, QP, nullptr, NRES, DMODEL, 192);
  gemm_wmma_kernel<<<dim3(NRES/128, 192/64), b256, 0, stream>>>(
      X, DMODEL, wkp, 192, nullptr, nullptr, KP, nullptr, NRES, DMODEL, 192);
  gemm_wmma_kernel<<<dim3(NRES/128, 192/64), b256, 0, stream>>>(
      X, DMODEL, wvp, 192, nullptr, nullptr, VP, nullptr, NRES, DMODEL, 192);
  int npt = NRES * NH * NPTS;
  ptglob_kernel<<<dim3((npt + 255) / 256), b256, 0, stream>>>(R, pos, QP);
  ptglob_kernel<<<dim3((npt + 255) / 256), b256, 0, stream>>>(R, pos, KP);
  ptglob_kernel<<<dim3((npt + 255) / 256), b256, 0, stream>>>(R, pos, VP);
  attn_kernel<<<dim3(NRES), b256, 0, stream>>>(Qb, Kb, Vb, QP, KP, VP, PAIR, wb, gamma,
                                               neigh, batch, mask, R, pos, IPA);
  gemm_wmma_kernel<<<dim3(NRES/128, DMODEL/64), b256, 0, stream>>>(
      IPA, FOUTC, wo, DMODEL, bo, ACC, ACC, nullptr, NRES, FOUTC, DMODEL);
  ln_kernel<<<dim3(NRES/8), b256, 0, stream>>>(ACC, ln2_s, ln2_o, X);
  gemm_wmma_kernel<<<dim3(NRES/128, DFF/64), b256, 0, stream>>>(
      X, DMODEL, wg, DFF, nullptr, nullptr, H1, nullptr, NRES, DMODEL, DFF);
  gemm_wmma_kernel<<<dim3(NRES/128, DFF/64), b256, 0, stream>>>(
      X, DMODEL, wvm, DFF, nullptr, nullptr, H2, nullptr, NRES, DMODEL, DFF);
  gelumul_kernel<<<dim3((NRES * DFF + 255) / 256), b256, 0, stream>>>(H1, H2, NRES * DFF);
  gemm_wmma_kernel<<<dim3(NRES/128, DMODEL/64), b256, 0, stream>>>(
      H1, DFF, wom, DMODEL, nullptr, ACC, ACC, out_local, NRES, DFF, DMODEL);
  ln_kernel<<<dim3(NRES/8), b256, 0, stream>>>(ACC, ln3_s, ln3_o, X);
  posupd_kernel<<<dim3((NRES + 255) / 256), b256, 0, stream>>>(X, wpos, LP, R, pos, umask, out_pos);
}
